// Attention_38225208934674
// MI455X (gfx1250) — compile-verified, matrix operand order checked by reading, not run on hardware
//
#include <hip/hip_runtime.h>
#include <hip/hip_bf16.h>
#include <math.h>

typedef __attribute__((ext_vector_type(16))) _Float16 v16h;
typedef __attribute__((ext_vector_type(8)))  _Float16 v8h;
typedef __attribute__((ext_vector_type(8)))  float    v8f;

#define DIMIN   40
#define EADc    16
#define HIDc    64
#define NWK     288
#define NWV     576
#define NTILE_K 18      // 288/16
#define NTILES  54      // (288+576)/16

// CDNA5 16-bit operand K mapping: lanes 0-15 hold K {0..7,16..23},
// lanes 16-31 hold K {8..15,24..31} across the 16 packed halves.
__device__ __forceinline__ int kmap(int lane, int j) {
  int base = (lane & 16) ? 8 : 0;
  return (j < 8) ? (base + j) : (base + 16 + (j - 8));
}

__device__ __forceinline__ void atomicMaxF(float* addr, float val) {
  if (val >= 0.0f) atomicMax((int*)addr, __float_as_int(val));
  else atomicMin((unsigned int*)addr, (unsigned int)__float_as_int(val));
}

__global__ void initk(float* amax, float* denom, float* out, int N) {
  int i = blockIdx.x * blockDim.x + threadIdx.x;
  if (i < N) { ((int*)amax)[i] = 0xFF800000; denom[i] = 0.0f; }
  if (i < N * 40) out[i] = 0.0f;
}

// Pre-swizzle W1 (16x128: W1k||W1v, K padded to 32) and W2 (64x864: W2k||W2v)
// into f16 B-operand fragments: layout [tile][kblk][lane][16 halves].
__global__ void prepw(const float* __restrict__ W1k, const float* __restrict__ W1v,
                      const float* __restrict__ W2k, const float* __restrict__ W2v,
                      _Float16* __restrict__ w1f, _Float16* __restrict__ w2f) {
  int idx = blockIdx.x * blockDim.x + threadIdx.x;
  if (idx < 4096) {                       // 8 tiles * 32 lanes * 16
    int tile = idx >> 9, rem = idx & 511;
    int lane = rem >> 4, j = rem & 15;
    int k = kmap(lane, j);
    int col = tile * 16 + (lane & 15);
    float v = 0.0f;
    if (k < 16) v = (col < 64) ? W1k[k * 64 + col] : W1v[k * 64 + (col - 64)];
    w1f[idx] = (_Float16)v;
  }
  int idx2 = idx - 4096;
  if (idx2 >= 0 && idx2 < 55296) {        // 54 tiles * 2 kblks * 32 lanes * 16
    int tile = idx2 >> 10, rem = idx2 & 1023;
    int kb = rem >> 9, r2 = rem & 511;
    int lane = r2 >> 4, j = r2 & 15;
    int k = kb * 32 + kmap(lane, j);
    int n = lane & 15;
    float v = (tile < NTILE_K) ? W2k[k * NWK + tile * 16 + n]
                               : W2v[k * NWV + (tile - NTILE_K) * 16 + n];
    w2f[idx2] = (_Float16)v;
  }
}

__global__ void qkern(const float* __restrict__ node_attr,
                      const float* __restrict__ Wq0, const float* __restrict__ Wq1,
                      float* __restrict__ q0, float* __restrict__ q1, int N) {
  int n = blockIdx.x * blockDim.x + threadIdx.x;
  if (n >= N) return;
  const float* x = node_attr + (size_t)n * DIMIN;
  for (int q = 0; q < 8; ++q) {
    float s = 0.0f;
    for (int m = 0; m < 16; ++m) s += x[m] * Wq0[m * 8 + q];
    q0[n * 8 + q] = s * 0.25f;                      // / sqrt(16)
  }
  const float r8 = 0.35355339059327373f;            // 1/sqrt(8)
  for (int q = 0; q < 4; ++q)
    for (int i = 0; i < 3; ++i) {
      float s = 0.0f;
      for (int m = 0; m < 8; ++m) s += x[16 + m * 3 + i] * Wq1[m * 4 + q];
      q1[n * 12 + q * 3 + i] = s * r8;
    }
}

__global__ void __launch_bounds__(256)
edge_pass1(const float* __restrict__ node_attr, const float* __restrict__ edge_attr,
           const float* __restrict__ edge_sh,
           const float* __restrict__ b1k, const float* __restrict__ b1v,
           const float* __restrict__ b2k, const float* __restrict__ b2v,
           const float* __restrict__ Wd0, const float* __restrict__ Wd1,
           const int* __restrict__ edge_index,
           const float* __restrict__ q0g, const float* __restrict__ q1g,
           const _Float16* __restrict__ w1f, const _Float16* __restrict__ w2f,
           float* __restrict__ abuf, float* __restrict__ vbuf,
           float* __restrict__ amax, int E) {
  __shared__ _Float16 eattr[16][16];       // edge_attr tile (f16 A operand source)
  __shared__ _Float16 hid[16][128];        // silu(hidden), f16, k-half || v-half
  __shared__ float f00[16][16], f01[16][16], f11[16][8], f10[16][8][3];
  __shared__ float sh1l[16][3];
  __shared__ float q0d[16][8], q1d[16][12];
  __shared__ int   srcl[16], dstl[16];
  __shared__ float accK0[16][8], accKT[16][4], accK1[16][12];
  __shared__ float accV0[16][16], accVT[16][8], accV1[16][24];

  const int tid  = threadIdx.x;
  const int lane = tid & 31;               // wave32
  const int wave = tid >> 5;               // 8 waves
  const int ebase = blockIdx.x * 16;

  for (int i = tid; i < 16 * 8;  i += 256) ((float*)accK0)[i] = 0.0f;
  for (int i = tid; i < 16 * 4;  i += 256) ((float*)accKT)[i] = 0.0f;
  for (int i = tid; i < 16 * 12; i += 256) ((float*)accK1)[i] = 0.0f;
  for (int i = tid; i < 16 * 16; i += 256) ((float*)accV0)[i] = 0.0f;
  for (int i = tid; i < 16 * 8;  i += 256) ((float*)accVT)[i] = 0.0f;
  for (int i = tid; i < 16 * 24; i += 256) ((float*)accV1)[i] = 0.0f;

  if (tid < 16) {
    int e = min(ebase + tid, E - 1);
    srcl[tid] = edge_index[e];
    dstl[tid] = edge_index[E + e];
    sh1l[tid][0] = edge_sh[e * 4 + 1];
    sh1l[tid][1] = edge_sh[e * 4 + 2];
    sh1l[tid][2] = edge_sh[e * 4 + 3];
  }
  __syncthreads();

  {
    int r = tid >> 4, c = tid & 15;
    int e = min(ebase + r, E - 1);
    eattr[r][c] = (_Float16)edge_attr[(size_t)e * EADc + c];
    float sh0 = edge_sh[e * 4 + 0];
    int s = srcl[r];
    float x0 = node_attr[(size_t)s * DIMIN + c];
    const float inv_s2 = 0.70710678118654752f;
    const float s00 = inv_s2 * 0.25f;                   // inv_s2 / sqrt(16)
    const float s11 = inv_s2 * 0.35355339059327373f;    // inv_s2 / sqrt(8)
    const float rsqrt3 = 0.57735026918962576f;
    f00[r][c] = x0 * sh0 * s00;
    f01[r][c] = x0 * s00;
    if (c < 8) {
      float x1a = node_attr[(size_t)s * DIMIN + 16 + c * 3 + 0];
      float x1b = node_attr[(size_t)s * DIMIN + 16 + c * 3 + 1];
      float x1c = node_attr[(size_t)s * DIMIN + 16 + c * 3 + 2];
      float dot = (x1a * sh1l[r][0] + x1b * sh1l[r][1] + x1c * sh1l[r][2]) * rsqrt3;
      f11[r][c] = dot * s11;
      float g = sh0 * s11;
      f10[r][c][0] = x1a * g;
      f10[r][c][1] = x1b * g;
      f10[r][c][2] = x1c * g;
    }
    int d = dstl[r];
    if (c < 8)  q0d[r][c] = q0g[d * 8 + c];
    if (c < 12) q1d[r][c] = q1g[d * 12 + c];
  }
  __syncthreads();

  const int n   = lane & 15;
  const int m   = lane & 15;
  const int b8  = (lane & 16) ? 8 : 0;     // per-lane K base
  const int hi8 = (lane & 16) ? 8 : 0;     // D-row base

  // ---- GEMM1: hid = silu(edge_attr @ [W1k||W1v] + b1), one 16x16 tile per wave
  {
    int col = wave * 16 + n;
    float bias = (col < 64) ? b1k[col] : b1v[col - 64];
    v8f c;
    for (int v = 0; v < 8; ++v) c[v] = bias;
    // A fragment: K = b8..b8+7 from edge_attr, K>=16 is zero pad.
    v8h lo = *(const v8h*)&eattr[m][b8];
    v8h zz = (v8h)(_Float16)0.0f;
    v16h a = __builtin_shufflevector(lo, zz, 0, 1, 2, 3, 4, 5, 6, 7,
                                             8, 9, 10, 11, 12, 13, 14, 15);
    v16h bfrag = *(const v16h*)(w1f + (size_t)wave * 512 + lane * 16);
    c = __builtin_amdgcn_wmma_f32_16x16x32_f16(false, a, false, bfrag,
                                               (short)0, c, false, false);
    for (int v = 0; v < 8; ++v) {
      float x = c[v];
      float sig = __builtin_amdgcn_rcpf(1.0f + __expf(-x));
      hid[v + hi8][col] = (_Float16)(x * sig);
    }
  }
  __syncthreads();

  // ---- GEMM2 (K=64, 2 WMMAs) fused with tensor-product contraction
  for (int t = wave; t < NTILES; t += 8) {
    bool vt = (t >= NTILE_K);
    int cbase = vt ? (t - NTILE_K) * 16 : t * 16;
    float bias = vt ? b2v[cbase + n] : b2k[cbase + n];
    v8f c;
    for (int v = 0; v < 8; ++v) c[v] = bias;
    int coff = vt ? 64 : 0;
    for (int kb = 0; kb < 2; ++kb) {
      // A fragment: two contiguous 8-half runs in the hid row.
      const _Float16* hrow = &hid[m][coff + kb * 32 + b8];
      v8h lo = *(const v8h*)(hrow);
      v8h hi = *(const v8h*)(hrow + 16);
      v16h a = __builtin_shufflevector(lo, hi, 0, 1, 2, 3, 4, 5, 6, 7,
                                               8, 9, 10, 11, 12, 13, 14, 15);
      v16h bfrag = *(const v16h*)(w2f + ((size_t)t * 2 + kb) * 512 + lane * 16);
      c = __builtin_amdgcn_wmma_f32_16x16x32_f16(false, a, false, bfrag,
                                                 (short)0, c, false, false);
    }
    int col = cbase + n;
    if (!vt) {                                       // k-path (288 cols)
      if (cbase < 128) {
        int mm = col >> 3, o = col & 7;
        for (int v = 0; v < 8; ++v) { int r = v + hi8; atomicAdd(&accK0[r][o], c[v] * f00[r][mm]); }
      } else if (cbase < 192) {
        int c2 = col - 128, mm = c2 >> 3, o = c2 & 7;
        for (int v = 0; v < 8; ++v) { int r = v + hi8; atomicAdd(&accK0[r][o], c[v] * f11[r][mm]); }
      } else if (cbase < 256) {
        int c2 = col - 192, mm = c2 >> 2, o = c2 & 3;
        for (int v = 0; v < 8; ++v) { int r = v + hi8; atomicAdd(&accKT[r][o], c[v] * f01[r][mm]); }
      } else {
        int c2 = col - 256, mm = c2 >> 2, o = c2 & 3;
        for (int v = 0; v < 8; ++v) {
          int r = v + hi8; float d = c[v];
          atomicAdd(&accK1[r][o * 3 + 0], d * f10[r][mm][0]);
          atomicAdd(&accK1[r][o * 3 + 1], d * f10[r][mm][1]);
          atomicAdd(&accK1[r][o * 3 + 2], d * f10[r][mm][2]);
        }
      }
    } else {                                         // v-path (576 cols)
      if (cbase < 256) {
        int mm = col >> 4, o = col & 15;
        for (int v = 0; v < 8; ++v) { int r = v + hi8; atomicAdd(&accV0[r][o], c[v] * f00[r][mm]); }
      } else if (cbase < 384) {
        int c2 = col - 256, mm = c2 >> 4, o = c2 & 15;
        for (int v = 0; v < 8; ++v) { int r = v + hi8; atomicAdd(&accV0[r][o], c[v] * f11[r][mm]); }
      } else if (cbase < 512) {
        int c2 = col - 384, mm = c2 >> 3, o = c2 & 7;
        for (int v = 0; v < 8; ++v) { int r = v + hi8; atomicAdd(&accVT[r][o], c[v] * f01[r][mm]); }
      } else {
        int c2 = col - 512, mm = c2 >> 3, o = c2 & 7;
        for (int v = 0; v < 8; ++v) {
          int r = v + hi8; float d = c[v];
          atomicAdd(&accV1[r][o * 3 + 0], d * f10[r][mm][0]);
          atomicAdd(&accV1[r][o * 3 + 1], d * f10[r][mm][1]);
          atomicAdd(&accV1[r][o * 3 + 2], d * f10[r][mm][2]);
        }
      }
    }
  }
  __syncthreads();

  // ---- per-edge finalize: logits + value store + segment max
  if (tid < 16) {
    int e = ebase + tid;
    if (e < E) {
      int r = tid;
      float k0v[8], k1v[12];
      for (int o = 0; o < 8; ++o) k0v[o] = accK0[r][o];
      for (int o = 0; o < 4; ++o)
        for (int i = 0; i < 3; ++i)
          k1v[o * 3 + i] = accKT[r][o] * sh1l[r][i] + accK1[r][o * 3 + i];
      float a0 = 0.0f;
      for (int q = 0; q < 8; ++q) {
        float qq = q0d[r][q];
        for (int k = 0; k < 8; ++k) a0 += qq * Wd0[q * 8 + k] * k0v[k];
      }
      float a1 = 0.0f;
      for (int q = 0; q < 4; ++q)
        for (int k = 0; k < 4; ++k) {
          float w = Wd1[q * 4 + k];
          for (int i = 0; i < 3; ++i) a1 += q1d[r][q * 3 + i] * w * k1v[k * 3 + i];
        }
      const float rsqrt3  = 0.57735026918962576f;
      const float rsqrt80 = 0.11180339887498949f;
      float a = (a0 + a1 * rsqrt3) * rsqrt80;
      abuf[e] = a;
      atomicMaxF(&amax[dstl[r]], a);
      float* vo = vbuf + (size_t)e * 40;
      for (int o = 0; o < 16; ++o) vo[o] = accV0[r][o];
      for (int o = 0; o < 8; ++o)
        for (int i = 0; i < 3; ++i)
          vo[16 + o * 3 + i] = accVT[r][o] * sh1l[r][i] + accV1[r][o * 3 + i];
    }
  }
}

__global__ void pass2(const int* __restrict__ edge_index, const float* __restrict__ amax,
                      float* __restrict__ abuf, float* __restrict__ denom, int E) {
  int e = blockIdx.x * blockDim.x + threadIdx.x;
  if (e >= E) return;
  int d = edge_index[E + e];
  float m = amax[d];
  if (!(m > -3.0e38f)) m = 0.0f;
  float ea = __expf(abuf[e] - m);
  abuf[e] = ea;
  atomicAdd(&denom[d], ea);
}

__global__ void pass3(const int* __restrict__ edge_index, const float* __restrict__ abuf,
                      const float* __restrict__ denom, const float* __restrict__ vbuf,
                      float* __restrict__ out, int E) {
  long idx = (long)blockIdx.x * blockDim.x + threadIdx.x;
  if (idx >= (long)E * 40) return;
  int e = (int)(idx / 40), j = (int)(idx % 40);
  int d = edge_index[E + e];
  float alpha = abuf[e] * __builtin_amdgcn_rcpf(fmaxf(denom[d], 1e-12f));
  atomicAdd(&out[d * 40 + j], alpha * vbuf[idx]);
}

extern "C" void kernel_launch(void* const* d_in, const int* in_sizes, int n_in,
                              void* d_out, int out_size, void* d_ws, size_t ws_size,
                              hipStream_t stream) {
  const float* node_attr = (const float*)d_in[0];
  const float* edge_attr = (const float*)d_in[1];
  const float* edge_sh   = (const float*)d_in[2];
  const float* Wq0 = (const float*)d_in[3];
  const float* Wq1 = (const float*)d_in[4];
  const float* W1k = (const float*)d_in[5];
  const float* b1k = (const float*)d_in[6];
  const float* W2k = (const float*)d_in[7];
  const float* b2k = (const float*)d_in[8];
  const float* W1v = (const float*)d_in[9];
  const float* b1v = (const float*)d_in[10];
  const float* W2v = (const float*)d_in[11];
  const float* b2v = (const float*)d_in[12];
  const float* Wd0 = (const float*)d_in[13];
  const float* Wd1 = (const float*)d_in[14];
  const int* edge_index = (const int*)d_in[15];
  const int N = in_sizes[0] / DIMIN;
  const int E = in_sizes[15] / 2;
  float* out = (float*)d_out;

  float* ws   = (float*)d_ws;
  float* q0   = ws;
  float* q1   = q0 + (size_t)N * 8;
  float* abuf = q1 + (size_t)N * 12;
  float* amax = abuf + (size_t)E;
  float* denom = amax + N;
  _Float16* w1f = (_Float16*)(denom + N);
  _Float16* w2f = w1f + 4096;
  float* vbuf = (float*)(w2f + 55296);

  hipLaunchKernelGGL(initk, dim3((unsigned)((N * 40 + 255) / 256)), dim3(256), 0, stream,
                     amax, denom, out, N);
  hipLaunchKernelGGL(prepw, dim3((4096 + 55296 + 255) / 256), dim3(256), 0, stream,
                     W1k, W1v, W2k, W2v, w1f, w2f);
  hipLaunchKernelGGL(qkern, dim3((unsigned)((N + 255) / 256)), dim3(256), 0, stream,
                     node_attr, Wq0, Wq1, q0, q1, N);
  hipLaunchKernelGGL(edge_pass1, dim3((unsigned)((E + 15) / 16)), dim3(256), 0, stream,
                     node_attr, edge_attr, edge_sh, b1k, b1v, b2k, b2v, Wd0, Wd1,
                     edge_index, q0, q1, w1f, w2f, abuf, vbuf, amax, E);
  hipLaunchKernelGGL(pass2, dim3((unsigned)((E + 255) / 256)), dim3(256), 0, stream,
                     edge_index, amax, abuf, denom, E);
  long tot = (long)E * 40;
  hipLaunchKernelGGL(pass3, dim3((unsigned)((tot + 255) / 256)), dim3(256), 0, stream,
                     edge_index, abuf, denom, vbuf, out, E);
}
